// GIN_MoleculeNet_10213432229965
// MI455X (gfx1250) — matrix operand order repaired and checked by reading, not compiled, not run on hardware
//
#include <hip/hip_runtime.h>
#include <hip/hip_bf16.h>

#define N_NODES  100000
#define N_EDGES  1600000
#define N_GRAPHS 2048
#define DIM      128
#define OUT_DIM  12
#define N_LAYERS 5

typedef __attribute__((ext_vector_type(16))) __bf16 v16bf;
typedef __attribute__((ext_vector_type(8)))  __bf16 v8bf;
typedef __attribute__((ext_vector_type(8)))  float  v8f;
typedef __attribute__((ext_vector_type(4)))  int    v4i;

#define LDS_STRIDE 136  // halves; 272B pitch -> rotates 4 banks per row

// gfx1250 async global->LDS path (guarded; falls back to sync LDS copy)
#if defined(__gfx1250__) && \
    __has_builtin(__builtin_amdgcn_global_load_async_to_lds_b128) && \
    __has_builtin(__builtin_amdgcn_s_wait_asynccnt)
#define USE_ASYNC_LDS 1
typedef __attribute__((address_space(1))) char glob_char;
typedef __attribute__((address_space(3))) char lds_char;
typedef __attribute__((address_space(1))) v4i  glob_v4i;
typedef __attribute__((address_space(3))) v4i  lds_v4i;
#else
#define USE_ASYNC_LDS 0
#endif

// ---------------------------------------------------------------------------
// WMMA helpers (CDNA5 16x16x32 bf16, f32 accumulate)
// ---------------------------------------------------------------------------
__device__ __forceinline__ v8f wmma_bf(v16bf a, v16bf b, v8f c) {
  // (neg_a, A, neg_b, B, c_mod, C, reuse_a, reuse_b)
  return __builtin_amdgcn_wmma_f32_16x16x32_bf16(false, a, false, b, (short)0, c,
                                                 false, false);
}

__device__ __forceinline__ v16bf a_frag_zero() {
  v16bf a;
#pragma unroll
  for (int i = 0; i < 16; ++i) a[i] = (__bf16)0.0f;
  return a;
}

// A-fragment (16-bit A 16x32 layout): lane l holds row M=(l&15);
// halves 0..7 = K c0..c0+7, halves 8..15 = K c0+16..c0+23,
// with c0 = kt*32 + (l>>4)*8.
// Builds bf16(transform(zprev) + agg) where transform = BN affine + ReLU
// (identity for layer 0).
__device__ __forceinline__ v16bf a_frag_in(const float* z, const float* agg,
                                           const float* sc, const float* sh,
                                           int do_bn, int row, int c0) {
  v16bf a;
  const float* pz = z   + (size_t)row * DIM + c0;
  const float* pa = agg + (size_t)row * DIM + c0;
  if (do_bn) {
#pragma unroll
    for (int i = 0; i < 8; ++i) {
      float hv = fmaxf(fmaf(pz[i], sc[c0 + i], sh[c0 + i]), 0.f);
      a[i] = (__bf16)(hv + pa[i]);
    }
#pragma unroll
    for (int i = 0; i < 8; ++i) {
      float hv = fmaxf(fmaf(pz[16 + i], sc[c0 + 16 + i], sh[c0 + 16 + i]), 0.f);
      a[8 + i] = (__bf16)(hv + pa[16 + i]);
    }
  } else {
#pragma unroll
    for (int i = 0; i < 8; ++i) a[i]     = (__bf16)(pz[i]      + pa[i]);
#pragma unroll
    for (int i = 0; i < 8; ++i) a[8 + i] = (__bf16)(pz[16 + i] + pa[16 + i]);
  }
  return a;
}

__device__ __forceinline__ v16bf a_frag_f32(const float* X, int row, int c0) {
  v16bf a;
  const float* p = X + (size_t)row * DIM + c0;
#pragma unroll
  for (int i = 0; i < 8; ++i) a[i]     = (__bf16)p[i];
#pragma unroll
  for (int i = 0; i < 8; ++i) a[8 + i] = (__bf16)p[16 + i];
  return a;
}

__device__ __forceinline__ v16bf a_frag_lds(const __bf16* L, int row, int c0) {
  v16bf a;
  const __bf16* p = L + row * LDS_STRIDE + c0;
#pragma unroll
  for (int i = 0; i < 8; ++i) a[i]     = p[i];
#pragma unroll
  for (int i = 0; i < 8; ++i) a[8 + i] = p[16 + i];
  return a;
}

// B-fragment from LDS-staged weights (padded pitch): lane l holds column
// N = n0 + (l&15); halves 0..7 = K kb..kb+7, 8..15 = K kb+16..kb+23, kb = kt*32 + (l>>4)*8.
__device__ __forceinline__ v16bf b_frag_lds(const __bf16* Lw, int k, int n0) {
  const int kb = (k & ~31) + ((k >> 4) & 1) * 8;
  const __bf16* p = Lw + kb * LDS_STRIDE + n0 + (k & 15);
  v16bf r;
#pragma unroll
  for (int i = 0; i < 8; ++i) { r[i] = p[i * LDS_STRIDE]; r[8 + i] = p[(16 + i) * LDS_STRIDE]; }
  return r;
}

// B-fragment straight from global (head kernel; tiny, 32B-aligned)
__device__ __forceinline__ v16bf b_frag(const __bf16* W, int ldw, int k, int n0) {
  const int kb = (k & ~31) + ((k >> 4) & 1) * 8;
  const __bf16* p = W + (size_t)kb * ldw + n0 + (k & 15);
  v16bf r;
#pragma unroll
  for (int i = 0; i < 8; ++i) { r[i] = p[(size_t)i * ldw]; r[8 + i] = p[(size_t)(16 + i) * ldw]; }
  return r;
}

// Stage a 128x128 bf16 weight matrix into LDS with padded pitch.
// 2048 16B chunks / 256 threads = 8 chunks per thread.
__device__ __forceinline__ void stage_weight_tile(const __bf16* __restrict__ Wg,
                                                  __bf16* Lw, int tid) {
#if USE_ASYNC_LDS
  glob_char* g = (glob_char*)Wg;
  lds_char*  l = (lds_char*)Lw;
#pragma unroll
  for (int i = 0; i < 8; ++i) {
    int chunk = tid + i * 256;
    int row = chunk >> 4, seg = chunk & 15;
    __builtin_amdgcn_global_load_async_to_lds_b128(
        (glob_v4i*)(g + row * (DIM * 2) + seg * 16),
        (lds_v4i*)(l + row * (LDS_STRIDE * 2) + seg * 16), 0, 0);
  }
#else
#pragma unroll
  for (int i = 0; i < 8; ++i) {
    int chunk = tid + i * 256;
    int row = chunk >> 4, seg = chunk & 15;
    *(v8bf*)(Lw + row * LDS_STRIDE + seg * 8) =
        *(const v8bf*)(Wg + row * DIM + seg * 8);
  }
#endif
}

__device__ __forceinline__ void stage_wait() {
#if USE_ASYNC_LDS
  __builtin_amdgcn_s_wait_asynccnt(0);
#endif
}

// ---------------------------------------------------------------------------
// Utility kernels
// ---------------------------------------------------------------------------
__global__ void zero_f32_kernel(float* __restrict__ p, int n) {
  int i = blockIdx.x * blockDim.x + threadIdx.x;
  if (i < n) p[i] = 0.0f;
}

__global__ void cvt_bf16_kernel(const float* __restrict__ in, __bf16* __restrict__ out,
                                int n) {
  int i = blockIdx.x * blockDim.x + threadIdx.x;
  if (i < n) out[i] = (__bf16)in[i];
}

__global__ void pad_wh2_kernel(const float* __restrict__ w, __bf16* __restrict__ o) {
  int t = blockIdx.x * blockDim.x + threadIdx.x;
  if (t >= DIM * 16) return;
  int k = t >> 4, n2 = t & 15;
  o[t] = (__bf16)(n2 < OUT_DIM ? w[k * OUT_DIM + n2] : 0.0f);
}

// ---------------------------------------------------------------------------
// Edge scatter with fused BN+ReLU of the previous layer:
// agg[dst] += relu(zprev[src]*sc + sh)     (identity transform for layer 0)
// One wave per edge, float4 per lane.
// ---------------------------------------------------------------------------
__global__ __launch_bounds__(256) void scatter_bn_kernel(
    const float* __restrict__ zprev, const float* __restrict__ sc,
    const float* __restrict__ sh, int do_bn, const int* __restrict__ src,
    const int* __restrict__ dst, float* __restrict__ agg) {
  long long t = (long long)blockIdx.x * blockDim.x + threadIdx.x;
  int e    = (int)(t >> 5);
  int lane = (int)(t & 31);
  if (e >= N_EDGES) return;
  int s = src[e], d = dst[e];
  float4 v = *(const float4*)(zprev + (size_t)s * DIM + lane * 4);
  if (do_bn) {
    const float4 s4 = *(const float4*)(sc + lane * 4);
    const float4 h4 = *(const float4*)(sh + lane * 4);
    v.x = fmaxf(fmaf(v.x, s4.x, h4.x), 0.f);
    v.y = fmaxf(fmaf(v.y, s4.y, h4.y), 0.f);
    v.z = fmaxf(fmaf(v.z, s4.z, h4.z), 0.f);
    v.w = fmaxf(fmaf(v.w, s4.w, h4.w), 0.f);
  }
  float* p = agg + (size_t)d * DIM + lane * 4;
  atomicAdd(p + 0, v.x);
  atomicAdd(p + 1, v.y);
  atomicAdd(p + 2, v.z);
  atomicAdd(p + 3, v.w);
}

// ---------------------------------------------------------------------------
// Fused GIN MLP: z = ReLU((h+agg)@W1 + b1) @ W2 + b2 with h reconstructed
// on the fly from zprev (BN+ReLU affine), plus per-feature sum/sumsq for the
// next BatchNorm. Weights staged in LDS (async DMA where available).
// z overwrites agg in place (each wave reads only its own 16 rows first).
// ---------------------------------------------------------------------------
__global__ __launch_bounds__(256) void mlp_kernel(
    const float* __restrict__ zprev, const float* __restrict__ sc,
    const float* __restrict__ sh, int do_bn, float* aggz,
    const __bf16* __restrict__ W1, const float* __restrict__ b1,
    const __bf16* __restrict__ W2, const float* __restrict__ b2,
    float* __restrict__ ssum, float* __restrict__ ssq, int n_strips) {
  __shared__ __bf16 sW1[DIM * LDS_STRIDE];
  __shared__ __bf16 sW2[DIM * LDS_STRIDE];
  __shared__ __bf16 sInter[8][16 * LDS_STRIDE];

  const int tid  = threadIdx.x;
  const int wave = tid >> 5;
  const int lane = tid & 31;
  const int half = lane >> 4;
  const int nl   = lane & 15;
  const int strip = blockIdx.x * 8 + wave;
  const bool valid = strip < n_strips;
  const int m0  = strip * 16;
  const int row = m0 + nl;

  // Issue weight staging (async DMA overlaps the A-fragment loads below)
  stage_weight_tile(W1, sW1, tid);
  stage_weight_tile(W2, sW2, tid);

  // ---- GEMM1 A-fragments: bf16(relu(bn(zprev)) + agg) ----
  v16bf a1[4];
  if (valid) {
#pragma unroll
    for (int kt = 0; kt < 4; ++kt)
      a1[kt] = a_frag_in(zprev, aggz, sc, sh, do_bn, row, kt * 32 + half * 8);
  } else {
#pragma unroll
    for (int kt = 0; kt < 4; ++kt) a1[kt] = a_frag_zero();
  }

  stage_wait();
  __syncthreads();

  __bf16* L = sInter[wave];
#pragma unroll
  for (int j = 0; j < 8; ++j) {
    v8f c = {0.f, 0.f, 0.f, 0.f, 0.f, 0.f, 0.f, 0.f};
#pragma unroll
    for (int kt = 0; kt < 4; ++kt)
      c = wmma_bf(a1[kt], b_frag_lds(sW1, kt * 32 + lane, j * 16), c);
    const float bias = b1[j * 16 + nl];
#pragma unroll
    for (int v = 0; v < 8; ++v) {
      float y = c[v] + bias;
      y = y > 0.f ? y : 0.f;
      L[(v + half * 8) * LDS_STRIDE + j * 16 + nl] = (__bf16)y;
    }
  }
  __syncthreads();

  // ---- GEMM2: reload intermediate from LDS in A layout ----
  v16bf a2[4];
#pragma unroll
  for (int kt = 0; kt < 4; ++kt) a2[kt] = a_frag_lds(L, nl, kt * 32 + half * 8);

#pragma unroll
  for (int j = 0; j < 8; ++j) {
    v8f c = {0.f, 0.f, 0.f, 0.f, 0.f, 0.f, 0.f, 0.f};
#pragma unroll
    for (int kt = 0; kt < 4; ++kt)
      c = wmma_bf(a2[kt], b_frag_lds(sW2, kt * 32 + lane, j * 16), c);
    const float bias = b2[j * 16 + nl];
    float s = 0.f, sq = 0.f;
#pragma unroll
    for (int v = 0; v < 8; ++v) {
      float zv = c[v] + bias;
      s += zv;
      sq += zv * zv;
      if (valid) aggz[(size_t)(m0 + v + half * 8) * DIM + j * 16 + nl] = zv;
    }
    // fold lanes 16..31 (rows 8..15) into lanes 0..15 before the atomics
    s  += __shfl_xor(s, 16, 32);
    sq += __shfl_xor(sq, 16, 32);
    if (valid && half == 0) {
      atomicAdd(&ssum[j * 16 + nl], s);
      atomicAdd(&ssq[j * 16 + nl], sq);
    }
  }
}

// ---------------------------------------------------------------------------
// BatchNorm finalize: fold mean/var/gamma/beta into per-feature scale/shift
// ---------------------------------------------------------------------------
__global__ void bn_finalize_kernel(const float* __restrict__ ssum,
                                   const float* __restrict__ ssq,
                                   const float* __restrict__ gamma,
                                   const float* __restrict__ beta,
                                   float* __restrict__ scale,
                                   float* __restrict__ shift) {
  int f = threadIdx.x;
  const float inv_n = 1.0f / (float)N_NODES;
  float mean = ssum[f] * inv_n;
  float var  = ssq[f] * inv_n - mean * mean;
  float s    = gamma[f] * rsqrtf(var + 1e-5f);
  scale[f] = s;
  shift[f] = beta[f] - mean * s;
}

// ---------------------------------------------------------------------------
// Graph pooling with fused final BN+ReLU: g[batch[n]] += relu(z*sc+sh)
// ---------------------------------------------------------------------------
__global__ void pool_bn_kernel(const float* __restrict__ z,
                               const float* __restrict__ sc,
                               const float* __restrict__ sh,
                               const int* __restrict__ batch,
                               float* __restrict__ g) {
  int i = blockIdx.x * blockDim.x + threadIdx.x;
  if (i >= N_NODES * DIM) return;
  int node = i >> 7, f = i & (DIM - 1);
  float y = fmaxf(fmaf(z[i], sc[f], sh[f]), 0.f);
  atomicAdd(&g[(size_t)batch[node] * DIM + f], y);
}

// ---------------------------------------------------------------------------
// Head: out = ReLU(g@Wh1 + bh1) @ Wh2 + bh2   (2048 rows, N padded 12->16)
// ---------------------------------------------------------------------------
__global__ __launch_bounds__(256) void head_kernel(
    const float* __restrict__ g, const __bf16* __restrict__ Wh1,
    const float* __restrict__ bh1, const __bf16* __restrict__ Wh2p,
    const float* __restrict__ bh2, float* __restrict__ out) {
  __shared__ __bf16 lds[8][16 * LDS_STRIDE];
  const int wave = threadIdx.x >> 5;
  const int lane = threadIdx.x & 31;
  const int half = lane >> 4;
  const int nl   = lane & 15;
  const int m0   = (blockIdx.x * 8 + wave) * 16;
  const int row  = m0 + nl;

  v16bf a1[4];
#pragma unroll
  for (int kt = 0; kt < 4; ++kt) a1[kt] = a_frag_f32(g, row, kt * 32 + half * 8);

  __bf16* L = lds[wave];
#pragma unroll
  for (int j = 0; j < 8; ++j) {
    v8f c = {0.f, 0.f, 0.f, 0.f, 0.f, 0.f, 0.f, 0.f};
#pragma unroll
    for (int kt = 0; kt < 4; ++kt)
      c = wmma_bf(a1[kt], b_frag(Wh1, DIM, kt * 32 + lane, j * 16), c);
    const float bias = bh1[j * 16 + nl];
#pragma unroll
    for (int v = 0; v < 8; ++v) {
      float y = c[v] + bias;
      y = y > 0.f ? y : 0.f;
      L[(v + half * 8) * LDS_STRIDE + j * 16 + nl] = (__bf16)y;
    }
  }
  __syncthreads();

  v16bf a2[4];
#pragma unroll
  for (int kt = 0; kt < 4; ++kt) a2[kt] = a_frag_lds(L, nl, kt * 32 + half * 8);

  v8f c = {0.f, 0.f, 0.f, 0.f, 0.f, 0.f, 0.f, 0.f};
#pragma unroll
  for (int kt = 0; kt < 4; ++kt)
    c = wmma_bf(a2[kt], b_frag(Wh2p, 16, kt * 32 + lane, 0), c);

  if (nl < OUT_DIM) {
    const float bias = bh2[nl];
#pragma unroll
    for (int v = 0; v < 8; ++v)
      out[(size_t)(m0 + v + half * 8) * OUT_DIM + nl] = c[v] + bias;
  }
}

// ---------------------------------------------------------------------------
// Host launch
// ---------------------------------------------------------------------------
extern "C" void kernel_launch(void* const* d_in, const int* in_sizes, int n_in,
                              void* d_out, int out_size, void* d_ws, size_t ws_size,
                              hipStream_t stream) {
  const float* x      = (const float*)d_in[0];
  const int*   ei     = (const int*)d_in[1];
  const int*   batch  = (const int*)d_in[2];
  const float* W1s    = (const float*)d_in[3];
  const float* b1s    = (const float*)d_in[4];
  const float* W2s    = (const float*)d_in[5];
  const float* b2s    = (const float*)d_in[6];
  const float* gammas = (const float*)d_in[7];
  const float* betas  = (const float*)d_in[8];
  const float* Wh1    = (const float*)d_in[9];
  const float* bh1    = (const float*)d_in[10];
  const float* Wh2    = (const float*)d_in[11];
  const float* bh2    = (const float*)d_in[12];
  float* out = (float*)d_out;

  char* ws = (char*)d_ws;
  size_t off = 0;
  auto alloc = [&](size_t bytes) -> char* {
    char* p = ws + off;
    off += (bytes + 255) & ~(size_t)255;
    return p;
  };

  float*  zb0   = (float*)alloc((size_t)N_NODES * DIM * 4);
  float*  zb1   = (float*)alloc((size_t)N_NODES * DIM * 4);
  float*  g     = (float*)alloc((size_t)N_GRAPHS * DIM * 4);
  float*  stats = (float*)alloc((size_t)N_LAYERS * 2 * DIM * 4);
  float*  scsh  = (float*)alloc((size_t)N_LAYERS * 2 * DIM * 4);
  __bf16* W1b   = (__bf16*)alloc((size_t)N_LAYERS * DIM * DIM * 2);
  __bf16* W2b   = (__bf16*)alloc((size_t)N_LAYERS * DIM * DIM * 2);
  __bf16* Wh1b  = (__bf16*)alloc((size_t)DIM * DIM * 2);
  __bf16* Wh2p  = (__bf16*)alloc((size_t)DIM * 16 * 2);
  float*  zbuf[2] = {zb0, zb1};

  // Per-launch weight conversion + zero init
  int nw = N_LAYERS * DIM * DIM;
  cvt_bf16_kernel<<<(nw + 255) / 256, 256, 0, stream>>>(W1s, W1b, nw);
  cvt_bf16_kernel<<<(nw + 255) / 256, 256, 0, stream>>>(W2s, W2b, nw);
  cvt_bf16_kernel<<<(DIM * DIM + 255) / 256, 256, 0, stream>>>(Wh1, Wh1b, DIM * DIM);
  pad_wh2_kernel<<<(DIM * 16 + 255) / 256, 256, 0, stream>>>(Wh2, Wh2p);
  zero_f32_kernel<<<(N_LAYERS * 2 * DIM + 255) / 256, 256, 0, stream>>>(
      stats, N_LAYERS * 2 * DIM);
  zero_f32_kernel<<<(N_GRAPHS * DIM + 255) / 256, 256, 0, stream>>>(
      g, N_GRAPHS * DIM);

  const int n_elem   = N_NODES * DIM;
  const int n_strips = N_NODES / 16;  // 6250
  const long long scatter_threads = (long long)N_EDGES * 32;

  const float* zprev = x;
  for (int l = 0; l < N_LAYERS; ++l) {
    float* aggbuf = zbuf[l & 1];
    const float* sc = scsh + (l ? (l - 1) * 2 * DIM : 0);
    const float* sh = sc + DIM;
    zero_f32_kernel<<<(n_elem + 255) / 256, 256, 0, stream>>>(aggbuf, n_elem);
    scatter_bn_kernel<<<(unsigned)((scatter_threads + 255) / 256), 256, 0, stream>>>(
        zprev, sc, sh, l > 0, ei, ei + N_EDGES, aggbuf);
    mlp_kernel<<<(n_strips + 7) / 8, 256, 0, stream>>>(
        zprev, sc, sh, l > 0, aggbuf,
        W1b + (size_t)l * DIM * DIM, b1s + l * DIM,
        W2b + (size_t)l * DIM * DIM, b2s + l * DIM,
        stats + l * 2 * DIM, stats + l * 2 * DIM + DIM, n_strips);
    bn_finalize_kernel<<<1, DIM, 0, stream>>>(
        stats + l * 2 * DIM, stats + l * 2 * DIM + DIM,
        gammas + l * DIM, betas + l * DIM,
        scsh + l * 2 * DIM, scsh + l * 2 * DIM + DIM);
    zprev = aggbuf;
  }

  pool_bn_kernel<<<(n_elem + 255) / 256, 256, 0, stream>>>(
      zprev, scsh + 4 * 2 * DIM, scsh + 4 * 2 * DIM + DIM, batch, g);
  head_kernel<<<(N_GRAPHS / 16) / 8, 256, 0, stream>>>(g, Wh1b, bh1, Wh2p, bh2, out);
}
